// MPMC_net_50311246905636
// MI455X (gfx1250) — matrix operand order repaired and checked by reading, not compiled, not run on hardware
//
#include <hip/hip_runtime.h>
#include <hip/hip_bf16.h>
#include <math.h>
#include <stdint.h>

#define N_NODES   16384
#define N_EDGES   262144
#define NHID      128
#define NBATCH    16
#define NSAMP     1024
#define DIM       4
#define NLAYERS   3

typedef __attribute__((ext_vector_type(16))) _Float16 v16h;
typedef __attribute__((ext_vector_type(8)))  _Float16 v8h;
typedef __attribute__((ext_vector_type(8)))  float    v8f;
typedef uint32_t v4u __attribute__((ext_vector_type(4)));
typedef uint32_t v8u __attribute__((ext_vector_type(8)));

// A-fragment (16x32 f16, wave32): lane holds row M = lane%16.
// elements 0-15 : K = kbase + 2*hi8 + 0..15   (hi8 = 8*(lane/16))
__device__ __forceinline__ v16h a_frag(const _Float16* __restrict__ row, int kbase, int hi8) {
  v16h a = *(const v16h*)(row + kbase + 2 * hi8);
  return a;
}

// ---- Tensor Data Mover: 1-D bulk copy global(f16) -> LDS --------------------
// D# group0: [1:0]=count=1, [63:32]=lds_addr, [120:64]=global_addr, [127:126]=type=2
// D# group1: data_size=2B, tensor_dim0=tile_dim0=n (1-D tile), dims 1/2 unused.
__device__ __forceinline__ void tdm_load_1d(uint32_t lds_addr, const void* gptr, uint32_t n) {
  uint64_t ga = (uint64_t)(uintptr_t)gptr;
  v4u g0;
  g0.x = 1u;                                   // count=1 (valid user descriptor)
  g0.y = lds_addr;                             // LDS byte address
  g0.z = (uint32_t)ga;                         // global_addr[31:0]
  g0.w = (uint32_t)(ga >> 32) | (2u << 30);    // global_addr[56:32] | type=2
  v8u g1;
  g1[0] = (1u << 16);                          // workgroup_mask=0, data_size=1 (2 bytes)
  g1[1] = (n & 0xFFFFu) << 16;                 // tensor_dim0[15:0] -> bits[63:48]
  g1[2] = (n >> 16) | (1u << 16);              // tensor_dim0[31:16], tensor_dim1=1 (lo16)
  g1[3] = (n & 0xFFFFu) << 16;                 // tile_dim0 -> bits[127:112]
  g1[4] = 0;                                   // tile_dim1=0, tile_dim2=0 (unused)
  g1[5] = n;                                   // tensor_dim0_stride[31:0]
  g1[6] = 0;
  g1[7] = 0;
  asm volatile("tensor_load_to_lds %0, %1" :: "s"(g0), "s"(g1) : "memory");
}

// ---------------- weight prep: transpose + f32->f16 --------------------------
__global__ void mpmc_prep_weights(const float* __restrict__ m1w, const float* __restrict__ m2w,
                                  const float* __restrict__ u1w, const float* __restrict__ u2w,
                                  _Float16* __restrict__ m1t, _Float16* __restrict__ m2t,
                                  _Float16* __restrict__ u1t, _Float16* __restrict__ u2t) {
  int idx = blockIdx.x * blockDim.x + threadIdx.x;
  if (idx < NLAYERS * NHID * 2 * NHID) {          // [l][n][k], k in 0..255
    int l = idx / (NHID * 2 * NHID);
    int r = idx % (NHID * 2 * NHID);
    int n = r / (2 * NHID);
    int k = r % (2 * NHID);
    m1t[idx] = (_Float16)m1w[((size_t)l * 2 * NHID + k) * NHID + n];
    u1t[idx] = (_Float16)u1w[((size_t)l * 2 * NHID + k) * NHID + n];
  }
  if (idx < NLAYERS * NHID * NHID) {              // [l][n][k], k in 0..127
    int l = idx / (NHID * NHID);
    int r = idx % (NHID * NHID);
    int n = r / NHID;
    int k = r % NHID;
    m2t[idx] = (_Float16)m2w[((size_t)l * NHID + k) * NHID + n];
    u2t[idx] = (_Float16)u2w[((size_t)l * NHID + k) * NHID + n];
  }
}

// ---------------- encoder: h = X @ enc_w + b ---------------------------------
__global__ void mpmc_encoder(const float* __restrict__ X, const float* __restrict__ W,
                             const float* __restrict__ b, float* __restrict__ h,
                             _Float16* __restrict__ h16) {
  int idx = blockIdx.x * blockDim.x + threadIdx.x;  // n*128 + c
  int n = idx >> 7, c = idx & (NHID - 1);
  float acc = b[c];
#pragma unroll
  for (int k = 0; k < DIM; ++k) acc += X[n * DIM + k] * W[k * NHID + c];
  h[idx]   = acc;
  h16[idx] = (_Float16)acc;
}

// ---------------- edge message MLP + scatter-add aggregation -----------------
// wave = 16 edges; GEMM1: [16x256]x[256x128], GEMM2: [16x128]x[128x128]
// Weights staged to LDS once per block via the Tensor Data Mover.
__global__ void mpmc_edge_msg(const _Float16* __restrict__ h16,
                              const int* __restrict__ src, const int* __restrict__ dst,
                              const _Float16* __restrict__ w1t, const float* __restrict__ b1,
                              const _Float16* __restrict__ w2t, const float* __restrict__ b2,
                              float* __restrict__ agg) {
  __shared__ __align__(128) _Float16 lds_w1[NHID * 2 * NHID];   // 64 KB
  __shared__ __align__(128) _Float16 lds_w2[NHID * NHID];       // 32 KB
  __shared__ __align__(32)  _Float16 mtile[8][16 * NHID];       // 32 KB
  const int lane = threadIdx.x & 31;
  const int wave = threadIdx.x >> 5;

  if (threadIdx.x < 32) {   // wave 0 issues both TDM transfers
    tdm_load_1d((uint32_t)(uintptr_t)(void*)&lds_w1[0], (const void*)w1t, NHID * 2 * NHID);
    tdm_load_1d((uint32_t)(uintptr_t)(void*)&lds_w2[0], (const void*)w2t, NHID * NHID);
    __builtin_amdgcn_s_wait_tensorcnt(0);
  }
  __syncthreads();

  const int e0  = (blockIdx.x * 8 + wave) * 16;
  const int M   = lane & 15;
  const int hi  = lane >> 4;
  const int hi8 = hi * 8, hi16 = hi * 16;

  const int nd = dst[e0 + M];
  const int ns = src[e0 + M];
  const _Float16* rowd = h16 + (size_t)nd * NHID;
  const _Float16* rows = h16 + (size_t)ns * NHID;
  _Float16* mrow = &mtile[wave][0];

  // GEMM1: m = relu([h[dst], h[src]] @ W1 + b1)
#pragma unroll
  for (int nt = 0; nt < 8; ++nt) {
    float bv = b1[nt * 16 + M];
    v8f acc = {};
    const _Float16* wt = &lds_w1[(size_t)(nt * 16 + M) * (2 * NHID)];
#pragma unroll
    for (int kc = 0; kc < 8; ++kc) {
      const _Float16* arow = (kc < 4) ? rowd : rows;
      v16h a = a_frag(arow, (kc & 3) * 32, hi8);
      v16h b = *(const v16h*)(wt + kc * 32 + hi16);             // ds reads
      acc = __builtin_amdgcn_wmma_f32_16x16x32_f16(false, a, false, b, (short)0, acc, false, false);
    }
#pragma unroll
    for (int r = 0; r < 8; ++r) {
      float v = acc[r] + bv;
      v = v > 0.0f ? v : 0.0f;
      mrow[(r + hi8) * NHID + nt * 16 + M] = (_Float16)v;       // [M][N] layout
    }
  }
  __syncthreads();

  // GEMM2: m = relu(m @ W2 + b2); scatter-add into agg[dst]
#pragma unroll
  for (int nt = 0; nt < 8; ++nt) {
    float bv = b2[nt * 16 + M];
    v8f acc = {};
    const _Float16* wt = &lds_w2[(size_t)(nt * 16 + M) * NHID];
#pragma unroll
    for (int kc = 0; kc < 4; ++kc) {
      v16h a = a_frag(mrow + M * NHID, kc * 32, hi8);           // ds reads
      v16h b = *(const v16h*)(wt + kc * 32 + hi16);             // ds reads
      acc = __builtin_amdgcn_wmma_f32_16x16x32_f16(false, a, false, b, (short)0, acc, false, false);
    }
#pragma unroll
    for (int r = 0; r < 8; ++r) {
      float v = acc[r] + bv;
      v = v > 0.0f ? v : 0.0f;
      int node = __shfl(nd, r + hi8, 32);
      atomicAdd(&agg[(size_t)node * NHID + nt * 16 + M], v);
    }
  }
}

// ---------------- f32 -> f16 elementwise -------------------------------------
__global__ void mpmc_f32_to_f16(const float* __restrict__ in, _Float16* __restrict__ out) {
  int idx = blockIdx.x * blockDim.x + threadIdx.x;
  out[idx] = (_Float16)in[idx];
}

// ---------------- node update MLP: u = relu(relu([h,agg]@U1+b1)@U2+b2) -------
__global__ void mpmc_node_update(const _Float16* __restrict__ h16,
                                 const _Float16* __restrict__ agg16,
                                 const _Float16* __restrict__ w1t, const float* __restrict__ b1,
                                 const _Float16* __restrict__ w2t, const float* __restrict__ b2,
                                 float* __restrict__ u) {
  __shared__ __align__(128) _Float16 lds_w1[NHID * 2 * NHID];   // 64 KB
  __shared__ __align__(128) _Float16 lds_w2[NHID * NHID];       // 32 KB
  __shared__ __align__(32)  _Float16 mtile[8][16 * NHID];       // 32 KB
  const int lane = threadIdx.x & 31;
  const int wave = threadIdx.x >> 5;

  if (threadIdx.x < 32) {
    tdm_load_1d((uint32_t)(uintptr_t)(void*)&lds_w1[0], (const void*)w1t, NHID * 2 * NHID);
    tdm_load_1d((uint32_t)(uintptr_t)(void*)&lds_w2[0], (const void*)w2t, NHID * NHID);
    __builtin_amdgcn_s_wait_tensorcnt(0);
  }
  __syncthreads();

  const int n0  = (blockIdx.x * 8 + wave) * 16;
  const int M   = lane & 15;
  const int hi  = lane >> 4;
  const int hi8 = hi * 8, hi16 = hi * 16;

  const _Float16* rowh = h16   + (size_t)(n0 + M) * NHID;
  const _Float16* rowa = agg16 + (size_t)(n0 + M) * NHID;
  _Float16* mrow = &mtile[wave][0];

#pragma unroll
  for (int nt = 0; nt < 8; ++nt) {
    float bv = b1[nt * 16 + M];
    v8f acc = {};
    const _Float16* wt = &lds_w1[(size_t)(nt * 16 + M) * (2 * NHID)];
#pragma unroll
    for (int kc = 0; kc < 8; ++kc) {
      const _Float16* arow = (kc < 4) ? rowh : rowa;
      v16h a = a_frag(arow, (kc & 3) * 32, hi8);
      v16h b = *(const v16h*)(wt + kc * 32 + hi16);
      acc = __builtin_amdgcn_wmma_f32_16x16x32_f16(false, a, false, b, (short)0, acc, false, false);
    }
#pragma unroll
    for (int r = 0; r < 8; ++r) {
      float v = acc[r] + bv;
      v = v > 0.0f ? v : 0.0f;
      mrow[(r + hi8) * NHID + nt * 16 + M] = (_Float16)v;
    }
  }
  __syncthreads();

#pragma unroll
  for (int nt = 0; nt < 8; ++nt) {
    float bv = b2[nt * 16 + M];
    v8f acc = {};
    const _Float16* wt = &lds_w2[(size_t)(nt * 16 + M) * NHID];
#pragma unroll
    for (int kc = 0; kc < 4; ++kc) {
      v16h a = a_frag(mrow + M * NHID, kc * 32, hi8);
      v16h b = *(const v16h*)(wt + kc * 32 + hi16);
      acc = __builtin_amdgcn_wmma_f32_16x16x32_f16(false, a, false, b, (short)0, acc, false, false);
    }
#pragma unroll
    for (int r = 0; r < 8; ++r) {
      float v = acc[r] + bv;
      v = v > 0.0f ? v : 0.0f;
      u[(size_t)(n0 + r + hi8) * NHID + nt * 16 + M] = v;
    }
  }
}

// ---------------- per-graph per-channel instance norm ------------------------
__global__ void mpmc_inorm(const float* __restrict__ u, float* __restrict__ h,
                           _Float16* __restrict__ h16) {
  const int b = blockIdx.x;
  const int c = threadIdx.x;
  const float* up = u + (size_t)b * NSAMP * NHID + c;
  float s = 0.0f, s2 = 0.0f;
  for (int i = 0; i < NSAMP; ++i) { float v = up[(size_t)i * NHID]; s += v; s2 += v * v; }
  float mean = s * (1.0f / NSAMP);
  float var  = s2 * (1.0f / NSAMP) - mean * mean;
  float inv  = rsqrtf(var + 1e-5f);
  float*    hp  = h   + (size_t)b * NSAMP * NHID + c;
  _Float16* hp6 = h16 + (size_t)b * NSAMP * NHID + c;
  for (int i = 0; i < NSAMP; ++i) {
    float v = (up[(size_t)i * NHID] - mean) * inv;
    hp[(size_t)i * NHID]  = v;
    hp6[(size_t)i * NHID] = (_Float16)v;
  }
}

// ---------------- decoder: x = sigmoid(h @ dec_w + b) ------------------------
__global__ void mpmc_decoder(const float* __restrict__ h, const float* __restrict__ W,
                             const float* __restrict__ b, float* __restrict__ x) {
  int n = blockIdx.x * blockDim.x + threadIdx.x;
  float acc[DIM] = {b[0], b[1], b[2], b[3]};
  const float* hp = h + (size_t)n * NHID;
  for (int c = 0; c < NHID; ++c) {
    float v = hp[c];
#pragma unroll
    for (int d = 0; d < DIM; ++d) acc[d] += v * W[c * DIM + d];
  }
#pragma unroll
  for (int d = 0; d < DIM; ++d) x[n * DIM + d] = 1.0f / (1.0f + expf(-acc[d]));
}

// ---------------- L2 discrepancy loss (mean over batch) ----------------------
__global__ void mpmc_loss(const float* __restrict__ x, float* __restrict__ out) {
  __shared__ float xs[NSAMP * DIM];   // 16 KB
  __shared__ float red1[256], red2[256];
  const int b = blockIdx.x, tid = threadIdx.x;
  const float* xb = x + (size_t)b * NSAMP * DIM;
  for (int i = tid; i < NSAMP * DIM; i += 256) xs[i] = xb[i];
  __syncthreads();
  float s1 = 0.0f, s2 = 0.0f;
  for (int i = tid; i < NSAMP; i += 256) {
    float a0 = xs[i * 4 + 0], a1 = xs[i * 4 + 1], a2 = xs[i * 4 + 2], a3 = xs[i * 4 + 3];
    s1 += (1.0f - a0 * a0) * (1.0f - a1 * a1) * (1.0f - a2 * a2) * (1.0f - a3 * a3);
    for (int j = 0; j < NSAMP; ++j) {
      float p = (1.0f - fmaxf(a0, xs[j * 4 + 0])) * (1.0f - fmaxf(a1, xs[j * 4 + 1])) *
                (1.0f - fmaxf(a2, xs[j * 4 + 2])) * (1.0f - fmaxf(a3, xs[j * 4 + 3]));
      s2 += p;
    }
  }
  red1[tid] = s1; red2[tid] = s2;
  __syncthreads();
  for (int off = 128; off > 0; off >>= 1) {
    if (tid < off) { red1[tid] += red1[tid + off]; red2[tid] += red2[tid + off]; }
    __syncthreads();
  }
  if (tid == 0) {
    float disc = sqrtf((1.0f / 81.0f) - (0.125f / (float)NSAMP) * red1[0] +
                       red2[0] * (1.0f / ((float)NSAMP * (float)NSAMP)));
    atomicAdd(out, disc * (1.0f / (float)NBATCH));
  }
}

// -----------------------------------------------------------------------------
extern "C" void kernel_launch(void* const* d_in, const int* in_sizes, int n_in,
                              void* d_out, int out_size, void* d_ws, size_t ws_size,
                              hipStream_t stream) {
  (void)in_sizes; (void)n_in; (void)out_size; (void)ws_size;
  const float* X     = (const float*)d_in[0];
  const int*   ei    = (const int*)  d_in[1];
  const float* enc_w = (const float*)d_in[3];
  const float* enc_b = (const float*)d_in[4];
  const float* dec_w = (const float*)d_in[5];
  const float* dec_b = (const float*)d_in[6];
  const float* m1_w  = (const float*)d_in[7];
  const float* m1_b  = (const float*)d_in[8];
  const float* m2_w  = (const float*)d_in[9];
  const float* m2_b  = (const float*)d_in[10];
  const float* u1_w  = (const float*)d_in[11];
  const float* u1_b  = (const float*)d_in[12];
  const float* u2_w  = (const float*)d_in[13];
  const float* u2_b  = (const float*)d_in[14];
  const int* src = ei;               // edge_index[0]
  const int* dst = ei + N_EDGES;     // edge_index[1]

  char* ws = (char*)d_ws;
  float*    h     = (float*)   (ws + 0);          //  8 MB
  _Float16* h16   = (_Float16*)(ws + 8388608);    //  4 MB
  float*    agg   = (float*)   (ws + 12582912);   //  8 MB (reused as u)
  _Float16* agg16 = (_Float16*)(ws + 20971520);   //  4 MB
  _Float16* m1t   = (_Float16*)(ws + 25165824);
  _Float16* u1t   = (_Float16*)(ws + 25362432);
  _Float16* m2t   = (_Float16*)(ws + 25559040);
  _Float16* u2t   = (_Float16*)(ws + 25657344);

  float* out = (float*)d_out;
  float* x   = out + 1;

  hipMemsetAsync(out, 0, sizeof(float), stream);

  mpmc_prep_weights<<<(NLAYERS * NHID * 2 * NHID + 255) / 256, 256, 0, stream>>>(
      m1_w, m2_w, u1_w, u2_w, m1t, m2t, u1t, u2t);
  mpmc_encoder<<<(N_NODES * NHID) / 256, 256, 0, stream>>>(X, enc_w, enc_b, h, h16);

  for (int l = 0; l < NLAYERS; ++l) {
    hipMemsetAsync(agg, 0, (size_t)N_NODES * NHID * sizeof(float), stream);
    mpmc_edge_msg<<<N_EDGES / (16 * 8), 256, 0, stream>>>(
        h16, src, dst,
        m1t + (size_t)l * NHID * 2 * NHID, m1_b + l * NHID,
        m2t + (size_t)l * NHID * NHID,     m2_b + l * NHID, agg);
    mpmc_f32_to_f16<<<(N_NODES * NHID) / 256, 256, 0, stream>>>(agg, agg16);
    mpmc_node_update<<<N_NODES / (16 * 8), 256, 0, stream>>>(
        h16, agg16,
        u1t + (size_t)l * NHID * 2 * NHID, u1_b + l * NHID,
        u2t + (size_t)l * NHID * NHID,     u2_b + l * NHID, agg /* u */);
    mpmc_inorm<<<NBATCH, NHID, 0, stream>>>(agg, h, h16);
  }

  mpmc_decoder<<<N_NODES / 256, 256, 0, stream>>>(h, dec_w, dec_b, x);
  mpmc_loss<<<NBATCH, 256, 0, stream>>>(x, out);
}
